// GlobalColAttention_70660801953921
// MI455X (gfx1250) — matrix operand order repaired and checked by reading, not compiled, not run on hardware
//
#include <hip/hip_runtime.h>
#include <hip/hip_bf16.h>
#include <math.h>

// ---------------------------------------------------------------------------
// GlobalColAttention for MI455X (gfx1250, wave32, WMMA).
// Precision: bf16 GEMM operands, f32 accumulate (v_wmma_f32_16x16x32_bf16).
// - Weights pre-transposed: B fragment = one contiguous 32B load (2x b128).
// - A fragments: two contiguous ds_load_b128 from the LDS row tile.
// - X tiles DMA'd global->LDS with GLOBAL_LOAD_ASYNC_TO_LDS_B128 (ASYNCcnt).
// - 32 rows per wave (two 16-row slabs) so each B fragment feeds 2 WMMAs,
//   halving L2 weight re-read traffic.
// ---------------------------------------------------------------------------

typedef __attribute__((ext_vector_type(16))) __bf16 v16bf;
typedef __attribute__((ext_vector_type(8)))  __bf16 v8bf;
typedef __attribute__((ext_vector_type(8)))  float  v8f;

#define S_DIM   1024
#define R_DIM   256
#define C_DIM   256      // c_em
#define HD_DIM  256      // h*d = 8*32
#define KV_DIM  64       // [K|V] concat
#define WROWS   32       // rows per wave (two 16-row WMMA slabs)
#define XPAD    264      // LDS row stride (bf16 elems), 528B = 33*16B
#define LN_EPS  1e-5f
#define QSCALE  0.17677669529663689f  // 1/sqrt(32)

// ---------------- async global -> LDS copy (CDNA5 TDM-lite path) ------------

__device__ inline unsigned lds_addr32(const void* p) {
    // low 32 bits of the flat address of an LDS object = LDS byte address
    return (unsigned)(size_t)p;
}

__device__ inline void async_b128_to_lds(unsigned ldsDst, const void* gsrc) {
    asm volatile("global_load_async_to_lds_b128 %0, %1, off"
                 :: "v"(ldsDst), "v"((unsigned long long)(size_t)gsrc)
                 : "memory");
}

__device__ inline void wait_async0() {
    asm volatile("s_wait_asynccnt 0" ::: "memory");
}

// ---------------- WMMA fragment loaders (contiguous k = 16g + j order, same in A and B) ----------------

// A fragment: 16x32 bf16. lane L: M=L&15, g=L>>4. Lane data = one contiguous
// 16-elem K-run: [kc*32+16g, +16)  -> 2x ds_load_b128.
__device__ inline v16bf a_frag(const __bf16 x[][XPAD], int row, int kc, int grp) {
    int kb = kc * 32 + 16 * grp;
    v8bf lo = *(const v8bf*)&x[row][kb];
    v8bf hi = *(const v8bf*)&x[row][kb + 8];
    return __builtin_shufflevector(lo, hi, 0, 1, 2, 3, 4, 5, 6, 7,
                                   8, 9, 10, 11, 12, 13, 14, 15);
}

// B fragment from transposed weights Wt[N][K] (ldk = K = 256).
// lane L: N=L&15, g=L>>4; element j -> K = kc*32 + 16g + j (contiguous 16).
__device__ inline v16bf b_frag_t(const __bf16* __restrict__ Wt, int col,
                                 int kc, int grp) {
    return *(const v16bf*)(Wt + col * 256 + kc * 32 + 16 * grp);
}

// ---------------- kernel 0: weight conversion to transposed bf16 ------------

__global__ __launch_bounds__(256)
void k0_prep(const float* __restrict__ w_g, const float* __restrict__ w_o,
             const float* __restrict__ w_k, const float* __restrict__ w_v,
             __bf16* __restrict__ WgT,   // [hd][c]
             __bf16* __restrict__ WoT,   // [c][hd]
             __bf16* __restrict__ WkvT)  // [d64][c]
{
    int t = blockIdx.x * blockDim.x + threadIdx.x;
    int stride = gridDim.x * blockDim.x;
    for (int i = t; i < C_DIM * HD_DIM; i += stride) {
        int n = i >> 8, k = i & 255;
        WgT[i] = (__bf16)w_g[k * HD_DIM + n];   // w_g[c][hd] -> WgT[hd][c]
        WoT[i] = (__bf16)w_o[k * C_DIM + n];    // w_o[hd][c] -> WoT[c][hd]
    }
    for (int i = t; i < KV_DIM * C_DIM; i += stride) {
        int d = i >> 8, c = i & 255;
        float v = (d < 32) ? w_k[c * 32 + d] : w_v[c * 32 + (d - 32)];
        WkvT[i] = (__bf16)v;
    }
}

// ---------------- kernel 1: LayerNorm + masked pool + Q --------------------

__global__ __launch_bounds__(256)
void k1_ln_pool_q(const float* __restrict__ m, const float* __restrict__ ln_g,
                  const float* __restrict__ ln_b, const float* __restrict__ w_q,
                  const int* __restrict__ mask,
                  __bf16* __restrict__ X, float* __restrict__ Q) {
    int r = blockIdx.x;
    int t = threadIdx.x;
    int wave = t >> 5, lane = t & 31;

    __shared__ float pool[C_DIM];
    __shared__ float qp[C_DIM];
    __shared__ float cred[256];

    pool[t] = 0.f;
    __syncthreads();

    float g_c[8], b_c[8];
#pragma unroll
    for (int i = 0; i < 8; ++i) {
        g_c[i] = ln_g[lane + 32 * i];
        b_c[i] = ln_b[lane + 32 * i];
    }

    float pacc[8] = {0, 0, 0, 0, 0, 0, 0, 0};

    for (int sBase = 0; sBase < S_DIM; sBase += 8) {
        int s = sBase + wave;
        const float* row = m + ((long)s * R_DIM + r) * C_DIM;
        float v[8];
        float sum = 0.f;
#pragma unroll
        for (int i = 0; i < 8; ++i) { v[i] = row[lane + 32 * i]; sum += v[i]; }
#pragma unroll
        for (int off = 16; off; off >>= 1) sum += __shfl_xor(sum, off, 32);
        float mu = sum * (1.f / 256.f);
        float var = 0.f;
#pragma unroll
        for (int i = 0; i < 8; ++i) { float d = v[i] - mu; var += d * d; }
#pragma unroll
        for (int off = 16; off; off >>= 1) var += __shfl_xor(var, off, 32);
        float inv = rsqrtf(var * (1.f / 256.f) + LN_EPS);
        float maskf = (float)mask[s * R_DIM + r];
        __bf16* xrow = X + ((long)s * R_DIM + r) * C_DIM;
#pragma unroll
        for (int i = 0; i < 8; ++i) {
            float x = (v[i] - mu) * inv * g_c[i] + b_c[i];
            xrow[lane + 32 * i] = (__bf16)x;
            pacc[i] += x * maskf;
        }
    }
#pragma unroll
    for (int i = 0; i < 8; ++i) atomicAdd(&pool[lane + 32 * i], pacc[i]);

    float cpart = 0.f;
    for (int s = t; s < S_DIM; s += 256) cpart += (float)mask[s * R_DIM + r];
    cred[t] = cpart;
    __syncthreads();
    for (int off = 128; off; off >>= 1) {
        if (t < off) cred[t] += cred[t + off];
        __syncthreads();
    }
    float cnt = cred[0];
    qp[t] = pool[t] / (cnt + 1e-10f);
    __syncthreads();

    float acc = 0.f;
#pragma unroll 4
    for (int c = 0; c < C_DIM; ++c) acc += qp[c] * w_q[c * HD_DIM + t];
    Q[r * HD_DIM + t] = acc * QSCALE;
}

// ---------------- kernel 2: KV = X @ Wkv (WMMA, 32 rows/wave) ---------------

__global__ __launch_bounds__(32)
void k2_kv_gemm(const __bf16* __restrict__ X, const __bf16* __restrict__ WkvT,
                float* __restrict__ KV) {
    __shared__ __bf16 xs[WROWS][XPAD];
    int lane = threadIdx.x;
    long tileBase = (long)blockIdx.x * WROWS;

    // async DMA: one b128 per lane per row, global -> LDS (no VGPR staging)
    for (int mm = 0; mm < WROWS; ++mm) {
        unsigned dst = lds_addr32(&xs[mm][0]) + lane * 16;
        async_b128_to_lds(dst, X + (tileBase + mm) * C_DIM + lane * 8);
    }
    wait_async0();

    int row = lane & 15, grp = lane >> 4;

    // hoist A fragments for both 16-row slabs
    v16bf a0[8], a1[8];
#pragma unroll
    for (int kc = 0; kc < 8; ++kc) {
        a0[kc] = a_frag(xs, row, kc, grp);
        a1[kc] = a_frag(xs, row + 16, kc, grp);
    }

#pragma unroll 1
    for (int nt = 0; nt < KV_DIM / 16; ++nt) {
        v8f acc0 = {}, acc1 = {};
        int col = nt * 16 + row;
#pragma unroll
        for (int kc = 0; kc < C_DIM / 32; ++kc) {
            v16bf b = b_frag_t(WkvT, col, kc, grp);
            acc0 = __builtin_amdgcn_wmma_f32_16x16x32_bf16(
                false, a0[kc], false, b, (short)0, acc0, false, false);
            acc1 = __builtin_amdgcn_wmma_f32_16x16x32_bf16(
                false, a1[kc], false, b, (short)0, acc1, false, false);
        }
#pragma unroll
        for (int j = 0; j < 8; ++j) {
            int mrow = j + 8 * grp;
            KV[(tileBase + mrow) * KV_DIM + col]      = acc0[j];
            KV[(tileBase + mrow + 16) * KV_DIM + col] = acc1[j];
        }
    }
}

// ---------------- kernel 3: logits + softmax(s) + O -------------------------

__global__ __launch_bounds__(256)
void k3_attn(const float* __restrict__ KV, const float* __restrict__ Q,
             const int* __restrict__ mask, float* __restrict__ O) {
    int r = blockIdx.x;
    int t = threadIdx.x;

    __shared__ float qs[HD_DIM];
    __shared__ float lg[S_DIM][8];   // 32 KB
    __shared__ float red[32][8];
    __shared__ float mx[8], sm[8];

    qs[t] = Q[r * HD_DIM + t];
    __syncthreads();

    for (int idx = t; idx < S_DIM * 8; idx += 256) {
        int s = idx >> 3, h = idx & 7;
        const float* k = KV + ((long)s * R_DIM + r) * KV_DIM;
        float d = 0.f;
#pragma unroll
        for (int j = 0; j < 32; ++j) d += qs[h * 32 + j] * k[j];
        float bias = mask[s * R_DIM + r] ? 0.f : -1e9f;
        lg[s][h] = d + bias;
    }
    __syncthreads();

    int h = t & 7, ch = t >> 3;
    float pm = -INFINITY;
    for (int s = ch * 32; s < ch * 32 + 32; ++s) pm = fmaxf(pm, lg[s][h]);
    red[ch][h] = pm;
    __syncthreads();
    if (t < 8) {
        float mmx = -INFINITY;
        for (int c = 0; c < 32; ++c) mmx = fmaxf(mmx, red[c][t]);
        mx[t] = mmx;
    }
    __syncthreads();
    float ps = 0.f;
    for (int s = ch * 32; s < ch * 32 + 32; ++s) {
        float e = expf(lg[s][h] - mx[h]);
        lg[s][h] = e;
        ps += e;
    }
    red[ch][h] = ps;
    __syncthreads();
    if (t < 8) {
        float ss = 0.f;
        for (int c = 0; c < 32; ++c) ss += red[c][t];
        sm[t] = ss;
    }
    __syncthreads();

    int ho = t >> 5, dd = t & 31;
    float inv = 1.f / sm[ho];
    float acc = 0.f;
    for (int s = 0; s < S_DIM; ++s) {
        acc += lg[s][ho] * KV[((long)s * R_DIM + r) * KV_DIM + 32 + dd];
    }
    O[r * HD_DIM + ho * 32 + dd] = acc * inv;
}

// ---------------- kernel 4: out = (sigmoid(X@Wg + bg) * O[r]) @ Wo + bo -----

__global__ __launch_bounds__(32)
void k4_gate_out(const __bf16* __restrict__ X, const __bf16* __restrict__ WgT,
                 const __bf16* __restrict__ WoT, const float* __restrict__ O,
                 const float* __restrict__ b_g, const float* __restrict__ b_o,
                 float* __restrict__ out) {
    __shared__ __bf16 xs[WROWS][XPAD];
    __shared__ __bf16 ts[WROWS][XPAD];
    __shared__ int rIdx[WROWS];

    int lane = threadIdx.x;
    long tileBase = (long)blockIdx.x * WROWS;

    for (int mm = 0; mm < WROWS; ++mm) {
        unsigned dst = lds_addr32(&xs[mm][0]) + lane * 16;
        async_b128_to_lds(dst, X + (tileBase + mm) * C_DIM + lane * 8);
    }
    rIdx[lane] = (int)((tileBase + lane) % R_DIM);
    wait_async0();

    int row = lane & 15, grp = lane >> 4;

    // GEMM1: T = sigmoid(X @ Wg + bg) * O[r]
    {
        v16bf a0[8], a1[8];
#pragma unroll
        for (int kc = 0; kc < 8; ++kc) {
            a0[kc] = a_frag(xs, row, kc, grp);
            a1[kc] = a_frag(xs, row + 16, kc, grp);
        }

#pragma unroll 1
        for (int nt = 0; nt < HD_DIM / 16; ++nt) {
            v8f acc0 = {}, acc1 = {};
            int n = nt * 16 + row;
#pragma unroll
            for (int kc = 0; kc < C_DIM / 32; ++kc) {
                v16bf b = b_frag_t(WgT, n, kc, grp);
                acc0 = __builtin_amdgcn_wmma_f32_16x16x32_bf16(
                    false, a0[kc], false, b, (short)0, acc0, false, false);
                acc1 = __builtin_amdgcn_wmma_f32_16x16x32_bf16(
                    false, a1[kc], false, b, (short)0, acc1, false, false);
            }
            float bg = b_g[n];
#pragma unroll
            for (int j = 0; j < 8; ++j) {
                int mrow = j + 8 * grp;
                float g0 = 1.f / (1.f + expf(-(acc0[j] + bg)));
                float g1 = 1.f / (1.f + expf(-(acc1[j] + bg)));
                ts[mrow][n]      = (__bf16)(g0 * O[rIdx[mrow] * HD_DIM + n]);
                ts[mrow + 16][n] = (__bf16)(g1 * O[rIdx[mrow + 16] * HD_DIM + n]);
            }
        }
    }
    __syncthreads();

    // GEMM2: out = T @ Wo + bo
    {
        v16bf a0[8], a1[8];
#pragma unroll
        for (int kc = 0; kc < 8; ++kc) {
            a0[kc] = a_frag(ts, row, kc, grp);
            a1[kc] = a_frag(ts, row + 16, kc, grp);
        }

#pragma unroll 1
        for (int nt = 0; nt < C_DIM / 16; ++nt) {
            v8f acc0 = {}, acc1 = {};
            int n = nt * 16 + row;
#pragma unroll
            for (int kc = 0; kc < HD_DIM / 32; ++kc) {
                v16bf b = b_frag_t(WoT, n, kc, grp);
                acc0 = __builtin_amdgcn_wmma_f32_16x16x32_bf16(
                    false, a0[kc], false, b, (short)0, acc0, false, false);
                acc1 = __builtin_amdgcn_wmma_f32_16x16x32_bf16(
                    false, a1[kc], false, b, (short)0, acc1, false, false);
            }
            float bo = b_o[n];
#pragma unroll
            for (int j = 0; j < 8; ++j) {
                int mrow = j + 8 * grp;
                out[(tileBase + mrow) * C_DIM + n]      = acc0[j] + bo;
                out[(tileBase + mrow + 16) * C_DIM + n] = acc1[j] + bo;
            }
        }
    }
}

// ---------------- host launcher --------------------------------------------

extern "C" void kernel_launch(void* const* d_in, const int* in_sizes, int n_in,
                              void* d_out, int out_size, void* d_ws, size_t ws_size,
                              hipStream_t stream) {
    // input order: m, ln_g, ln_b, w_q, w_k, w_v, w_g, b_g, w_o, b_o, mask
    const float* m    = (const float*)d_in[0];
    const float* ln_g = (const float*)d_in[1];
    const float* ln_b = (const float*)d_in[2];
    const float* w_q  = (const float*)d_in[3];
    const float* w_k  = (const float*)d_in[4];
    const float* w_v  = (const float*)d_in[5];
    const float* w_g  = (const float*)d_in[6];
    const float* b_g  = (const float*)d_in[7];
    const float* w_o  = (const float*)d_in[8];
    const float* b_o  = (const float*)d_in[9];
    const int*   mask = (const int*)d_in[10];
    float* out = (float*)d_out;

    // workspace layout
    char* ws = (char*)d_ws;
    const size_t NP = (size_t)S_DIM * R_DIM;               // 262144 rows
    __bf16* X    = (__bf16*)ws;                            // 128 MB
    float*  KV   = (float*)(ws + NP * C_DIM * 2);          // 64 MB
    float*  Qm   = (float*)((char*)KV + NP * KV_DIM * 4);  // 256 KB
    float*  Om   = Qm + R_DIM * HD_DIM;                    // 256 KB
    __bf16* WgT  = (__bf16*)(Om + R_DIM * HD_DIM);         // 128 KB
    __bf16* WoT  = WgT + HD_DIM * C_DIM;                   // 128 KB
    __bf16* WkvT = WoT + C_DIM * HD_DIM;                   // 32 KB

    k0_prep<<<64, 256, 0, stream>>>(w_g, w_o, w_k, w_v, WgT, WoT, WkvT);
    k1_ln_pool_q<<<R_DIM, 256, 0, stream>>>(m, ln_g, ln_b, w_q, mask, X, Qm);
    k2_kv_gemm<<<(int)(NP / WROWS), 32, 0, stream>>>(X, WkvT, KV);
    k3_attn<<<R_DIM, 256, 0, stream>>>(KV, Qm, mask, Om);
    k4_gate_out<<<(int)(NP / WROWS), 32, 0, stream>>>(X, WgT, WoT, Om, b_g, b_o, out);

    (void)in_sizes; (void)n_in; (void)out_size; (void)ws_size;
}
